// ALIGNN_62869731279395
// MI455X (gfx1250) — compile-verified, matrix operand order checked by reading, not run on hardware
//
#include <hip/hip_runtime.h>
#include <hip/hip_bf16.h>

// ============================================================================
// ALIGNN forward for MI455X (gfx1250, wave32, WMMA).
// Dense linears (H=256) run through a bf16 WMMA GEMM (v_wmma_f32_16x16x32_bf16)
// with f32 accumulation; activations stay f32 in HBM (BN needs f32 stats) and
// are converted to bf16 while staging tiles into LDS (batched float4 global
// loads, packed bf16 LDS stores, double-buffered so tile k+1 global loads
// overlap the ds_load/wmma stream of tile k). Grid is column-major so the 4
// column-blocks sharing an A row-panel are adjacent in dispatch order and the
// panel comes from L2 instead of HBM 4x. Everything else (gather, sigmoid-
// gated scatter-add segment sums, batchnorm, SiLU, pooling) is float4-
// vectorized memory-bound code -- the whole net is HBM-bound (~20 GB moved
// per call vs 0.6 TFLOP of GEMM).
//
// Input indexing assumption (compile-only; cannot be runtime-verified):
//   top level in setup_inputs() insertion order:
//     0 atom_features, 1 r, 2 angle_h, [3..192] params leaves,
//     193 g_src, 194 g_dst, 195 lg_src, 196 lg_dst, 197 node_graph
//   params flattened as a pytree with dict keys in SORTED order, lists in
//   index order:
//     alignn[0].edge(14) alignn[0].node(14) ... alignn[3].node   -> 3..114
//     angle_emb1(4) 115  angle_emb2(4) 119  atom_emb(4) 123
//     edge_emb1(4) 127   edge_emb2(4) 131   fc.b 135 fc.w 136
//     gcn[0..3](14 each) 137..192
//   eggc leaf order (sorted keys): bn_edges.b, bn_edges.g, bn_nodes.b,
//     bn_nodes.g, dst_gate.b, dst_gate.w, dst_update.b, dst_update.w,
//     edge_gate.b, edge_gate.w, src_gate.b, src_gate.w, src_update.b,
//     src_update.w
//   mlp leaf order: bn.b, bn.g, lin.b, lin.w
//
// Workspace requirement: ~1.37 GB (see layout in kernel_launch).
// ============================================================================

#define NN 10000
#define NE 120000
#define NT 360000
#define NG 100
#define HH 256
#define C4 64  // HH/4

typedef __attribute__((ext_vector_type(16))) __bf16 v16bf;
typedef __attribute__((ext_vector_type(8)))  float  v8f;

// ---------------------------------------------------------------------------
// WMMA GEMM:  C[rows x Ncols] = A[rows x K] * W[K x Ncols] + bias
// Block tile 128x64, 8 waves, each wave 32x32 (2x2 WMMA 16x16x32 bf16 tiles).
// K must be a multiple of 4 (true for 256/92/80/64/40); Ncols a multiple of 64.
// Correctness of K-tail: only the W tile is zero-padded past K; A K-tail lanes
// load clamped (real, finite) data and multiply against zero W rows.
// ---------------------------------------------------------------------------
#define TM 128
#define TN 64
#define LSTR 40  // bf16 elems per LDS row: 32 data + 8 pad = 80 bytes

union FragBF { v16bf v; float4 f4[2]; };

__device__ __forceinline__ uint32_t pk_bf16(float lo, float hi) {
  union { __bf16 b[2]; uint32_t u; } p;
  p.b[0] = (__bf16)lo;
  p.b[1] = (__bf16)hi;
  return p.u;
}

// Stage one 128x32 A tile and one 32x64 W tile (transposed) into LDS.
// Phase 1 issues all 6 b128 global loads back-to-back; phase 2 converts and
// stores to LDS, so loadcnt waits overlap instead of serializing per load.
__device__ __forceinline__ void stage_tile(
    const float* __restrict__ A, const float* __restrict__ W,
    int rows, int K, int Ncols, int row0, int col0, int k0, int tid,
    __bf16* __restrict__ sA, __bf16* __restrict__ sB)
{
  const float4* __restrict__ A4 = reinterpret_cast<const float4*>(A);
  const float4* __restrict__ W4 = reinterpret_cast<const float4*>(W);
  const int K4 = K >> 2, N4 = Ncols >> 2;

  // ---- phase 1: loads ----
  float4 va[4];
#pragma unroll
  for (int i = 0; i < 4; ++i) {
    int idx = tid + i * 256;      // 0..1023 over 128 rows x 8 float4
    int rr = idx >> 3;
    int kq = idx & 7;
    int gr = row0 + rr;  if (gr >= rows) gr = rows - 1;   // clamp, no mask
    int gk = k0 + (kq << 2); if (gk >= K) gk = K - 4;     // clamp, no mask
    va[i] = A4[(size_t)gr * K4 + (gk >> 2)];
  }
  float4 vb[2];
  bool inb[2];
#pragma unroll
  for (int i = 0; i < 2; ++i) {
    int idx = tid + i * 256;      // 0..511 over 32 k-rows x 16 float4
    int kk = idx >> 4;
    int nq = idx & 15;
    int gk = k0 + kk;
    inb[i] = gk < K;
    if (gk >= K) gk = K - 1;
    vb[i] = W4[(size_t)gk * N4 + ((col0 >> 2) + nq)];
  }

  // ---- phase 2: convert + LDS stores ----
#pragma unroll
  for (int i = 0; i < 4; ++i) {
    int idx = tid + i * 256;
    int rr = idx >> 3;
    int kq = idx & 7;
    uint2 pv;
    pv.x = pk_bf16(va[i].x, va[i].y);
    pv.y = pk_bf16(va[i].z, va[i].w);
    *reinterpret_cast<uint2*>(&sA[rr * LSTR + (kq << 2)]) = pv;
  }
#pragma unroll
  for (int i = 0; i < 2; ++i) {
    int idx = tid + i * 256;
    int kk = idx >> 4;
    int nq = idx & 15;
    float4 v = vb[i];
    if (!inb[i]) { v.x = 0.f; v.y = 0.f; v.z = 0.f; v.w = 0.f; }
    int nn = nq << 2;
    sB[(nn + 0) * LSTR + kk] = (__bf16)v.x;
    sB[(nn + 1) * LSTR + kk] = (__bf16)v.y;
    sB[(nn + 2) * LSTR + kk] = (__bf16)v.z;
    sB[(nn + 3) * LSTR + kk] = (__bf16)v.w;
  }
}

__global__ __launch_bounds__(256) void k_gemm_bf16(
    const float* __restrict__ A, const float* __restrict__ W,
    const float* __restrict__ bias, float* __restrict__ Co,
    int rows, int K, int Ncols)
{
  __shared__ __attribute__((aligned(16))) __bf16 sA[2][TM * LSTR];
  __shared__ __attribute__((aligned(16))) __bf16 sB[2][TN * LSTR];

  // grid.x = column blocks (few), grid.y = row blocks (many): the col-blocks
  // sharing one A row-panel are adjacent in dispatch order -> A hits L2.
  const int row0 = blockIdx.y * TM;
  const int col0 = blockIdx.x * TN;
  const int tid  = threadIdx.x;
  const int lane = tid & 31;
  const int wave = tid >> 5;
  const int wrow = (wave & 3) * 32;   // 0,32,64,96
  const int wcol = (wave >> 2) * 32;  // 0,32

  v8f acc[2][2];
#pragma unroll
  for (int a = 0; a < 2; ++a)
#pragma unroll
    for (int b = 0; b < 2; ++b)
#pragma unroll
      for (int g = 0; g < 8; ++g) acc[a][b][g] = 0.0f;

  const int ksteps = (K + 31) >> 5;
  stage_tile(A, W, rows, K, Ncols, row0, col0, 0, tid, sA[0], sB[0]);
  __syncthreads();

  for (int kt = 0; kt < ksteps; ++kt) {
    const int cur = kt & 1;
    if (kt + 1 < ksteps)  // prefetch next tile into the other buffer
      stage_tile(A, W, rows, K, Ncols, row0, col0, (kt + 1) << 5, tid,
                 sA[1 - cur], sB[1 - cur]);

    // Fragment loads per documented 16-bit A layout (wave32):
    // lanes 0-15: m = lane,   K halves {0..7, 16..23}
    // lanes 16-31: m = lane-16, K halves {8..15, 24..31}
    const int am   = lane & 15;
    const int koff = (lane < 16) ? 0 : 8;
    FragBF a0, a1, b0, b1;
    {
      const __bf16* p = &sA[cur][(wrow + am) * LSTR + koff];
      a0.f4[0] = *reinterpret_cast<const float4*>(p);
      a0.f4[1] = *reinterpret_cast<const float4*>(p + 16);
      p = &sA[cur][(wrow + 16 + am) * LSTR + koff];
      a1.f4[0] = *reinterpret_cast<const float4*>(p);
      a1.f4[1] = *reinterpret_cast<const float4*>(p + 16);
      p = &sB[cur][(wcol + am) * LSTR + koff];
      b0.f4[0] = *reinterpret_cast<const float4*>(p);
      b0.f4[1] = *reinterpret_cast<const float4*>(p + 16);
      p = &sB[cur][(wcol + 16 + am) * LSTR + koff];
      b1.f4[0] = *reinterpret_cast<const float4*>(p);
      b1.f4[1] = *reinterpret_cast<const float4*>(p + 16);
    }
    acc[0][0] = __builtin_amdgcn_wmma_f32_16x16x32_bf16(
        false, a0.v, false, b0.v, (short)0, acc[0][0], false, false);
    acc[0][1] = __builtin_amdgcn_wmma_f32_16x16x32_bf16(
        false, a0.v, false, b1.v, (short)0, acc[0][1], false, false);
    acc[1][0] = __builtin_amdgcn_wmma_f32_16x16x32_bf16(
        false, a1.v, false, b0.v, (short)0, acc[1][0], false, false);
    acc[1][1] = __builtin_amdgcn_wmma_f32_16x16x32_bf16(
        false, a1.v, false, b1.v, (short)0, acc[1][1], false, false);
    __syncthreads();
  }

  // Epilogue. C/D layout: lanes 0-15 n=lane, m=g; lanes 16-31 n=lane-16, m=8+g.
  const int n  = lane & 15;
  const int mb = (lane >> 4) * 8;
  if (row0 + TM <= rows) {  // full block: no guards
#pragma unroll
    for (int ti = 0; ti < 2; ++ti)
#pragma unroll
      for (int tj = 0; tj < 2; ++tj) {
        int r0 = row0 + wrow + ti * 16 + mb;
        int c0 = col0 + wcol + tj * 16;
        float bq = bias ? bias[c0 + n] : 0.0f;
#pragma unroll
        for (int g = 0; g < 8; ++g)
          Co[(size_t)(r0 + g) * Ncols + c0 + n] = acc[ti][tj][g] + bq;
      }
  } else {
#pragma unroll
    for (int ti = 0; ti < 2; ++ti)
#pragma unroll
      for (int tj = 0; tj < 2; ++tj) {
        int r0 = row0 + wrow + ti * 16 + mb;
        int c0 = col0 + wcol + tj * 16;
        float bq = bias ? bias[c0 + n] : 0.0f;
#pragma unroll
        for (int g = 0; g < 8; ++g)
          if (r0 + g < rows)
            Co[(size_t)(r0 + g) * Ncols + c0 + n] = acc[ti][tj][g] + bq;
      }
  }
}

// ---------------------------------------------------------------------------
// Elementwise / graph kernels (H fixed: c4 = 64 -> shift/mask, no int div)
// ---------------------------------------------------------------------------
__global__ void k_zero(float* p, long n) {
  long i = blockIdx.x * (long)blockDim.x + threadIdx.x;
  if (i < n) p[i] = 0.0f;
}

// m[e] += A[src[e]] + B[dst[e]]
__global__ void k_gather_add(float4* __restrict__ m,
                             const float4* __restrict__ a,
                             const float4* __restrict__ b,
                             const int* __restrict__ src,
                             const int* __restrict__ dst, long ne)
{
  long i = blockIdx.x * (long)blockDim.x + threadIdx.x;
  if (i >= ne * C4) return;
  long e = i >> 6;
  int c = (int)(i & 63);
  float4 mv = m[i];
  float4 av = a[((size_t)src[e] << 6) + c];
  float4 bv = b[((size_t)dst[e] << 6) + c];
  mv.x += av.x + bv.x; mv.y += av.y + bv.y;
  mv.z += av.z + bv.z; mv.w += av.w + bv.w;
  m[i] = mv;
}

__device__ __forceinline__ float sigf(float x) {
  return 1.0f / (1.0f + __expf(-x));
}

// s = sigmoid(m[e]); atomically: sumsh[dst[e]] += Bh[src[e]]*s; sums[dst[e]] += s
__global__ void k_scatter_gate(const float4* __restrict__ m,
                               const float4* __restrict__ bh,
                               const int* __restrict__ src,
                               const int* __restrict__ dst,
                               float* __restrict__ sumsh,
                               float* __restrict__ sums, long ne)
{
  long i = blockIdx.x * (long)blockDim.x + threadIdx.x;
  if (i >= ne * C4) return;
  long e = i >> 6;
  int c = (int)(i & 63);
  float4 mv = m[i];
  float4 s; s.x = sigf(mv.x); s.y = sigf(mv.y); s.z = sigf(mv.z); s.w = sigf(mv.w);
  float4 bv = bh[((size_t)src[e] << 6) + c];
  size_t o = (((size_t)dst[e] << 6) + c) << 2;
  atomicAdd(&sumsh[o + 0], bv.x * s.x);
  atomicAdd(&sumsh[o + 1], bv.y * s.y);
  atomicAdd(&sumsh[o + 2], bv.z * s.z);
  atomicAdd(&sumsh[o + 3], bv.w * s.w);
  atomicAdd(&sums[o + 0], s.x);
  atomicAdd(&sums[o + 1], s.y);
  atomicAdd(&sums[o + 2], s.z);
  atomicAdd(&sums[o + 3], s.w);
}

// td += sumsh / (sums + 1e-6)
__global__ void k_combine(float* __restrict__ td, const float* __restrict__ sh,
                          const float* __restrict__ ss, long n)
{
  long i = blockIdx.x * (long)blockDim.x + threadIdx.x;
  if (i < n) td[i] += sh[i] / (ss[i] + 1e-6f);
}

// column-wise sum / sumsq via atomics; block = C threads, each block strides rows
__global__ void k_bn_stats(const float* __restrict__ t, long rows, int C,
                           float* __restrict__ stats)
{
  int c = threadIdx.x;
  float s = 0.0f, s2 = 0.0f;
  for (long r = blockIdx.x; r < rows; r += gridDim.x) {
    float v = t[r * C + c];
    s += v; s2 += v * v;
  }
  atomicAdd(&stats[c], s);
  atomicAdd(&stats[C + c], s2);
}

// dst = (res?res:0) + silu(gamma*(t-mu)*rsqrt(var+eps)+beta); C = cmask+1 (pow2)
__global__ void k_bn_act(const float* __restrict__ t, const float* __restrict__ res,
                         float* __restrict__ dst, const float* __restrict__ gamma,
                         const float* __restrict__ beta,
                         const float* __restrict__ stats, long rows, int cmask)
{
  long i = blockIdx.x * (long)blockDim.x + threadIdx.x;
  if (i >= rows * (long)(cmask + 1)) return;
  int c = (int)(i & cmask);
  float inv = 1.0f / (float)rows;
  float mu = stats[c] * inv;
  float var = stats[cmask + 1 + c] * inv - mu * mu;
  float xn = gamma[c] * (t[i] - mu) * rsqrtf(var + 1e-5f) + beta[c];
  float sl = xn * sigf(xn);
  dst[i] = (res ? res[i] : 0.0f) + sl;
}

// RBF of bond length: out[e][k] = exp(-79/8 * (|r_e| - 8k/79)^2), 80 bins
__global__ void k_edge_rbf(const float* __restrict__ r, float* __restrict__ out, int E)
{
  int e = blockIdx.x * blockDim.x + threadIdx.x;
  if (e >= E) return;
  float x = r[3 * e + 0], y = r[3 * e + 1], z = r[3 * e + 2];
  float d = sqrtf(x * x + y * y + z * z);
  const float gamma = 79.0f / 8.0f;
#pragma unroll 8
  for (int k = 0; k < 80; ++k) {
    float ck = 8.0f * (float)k / 79.0f;
    float t = d - ck;
    out[(size_t)e * 80 + k] = __expf(-gamma * t * t);
  }
}

// RBF of angle_h: 40 bins over [-1,1], gamma = 39/2
__global__ void k_angle_rbf(const float* __restrict__ a, float* __restrict__ out, int Tn)
{
  int e = blockIdx.x * blockDim.x + threadIdx.x;
  if (e >= Tn) return;
  float d = a[e];
  const float gamma = 39.0f / 2.0f;
#pragma unroll 8
  for (int k = 0; k < 40; ++k) {
    float ck = -1.0f + 2.0f * (float)k / 39.0f;
    float t = d - ck;
    out[(size_t)e * 40 + k] = __expf(-gamma * t * t);
  }
}

// mean-pool scatter: gout[ng[n]] += x[n]; gcnt[ng[n]] += 1
__global__ void k_pool(const float4* __restrict__ x, const int* __restrict__ ng,
                       float* __restrict__ gout, float* __restrict__ gcnt, int Nn)
{
  long i = blockIdx.x * (long)blockDim.x + threadIdx.x;
  if (i >= (long)Nn * C4) return;
  long n = i >> 6;
  int c = (int)(i & 63);
  int g = ng[n];
  float4 v = x[i];
  size_t o = (((size_t)g << 6) + c) << 2;
  atomicAdd(&gout[o + 0], v.x);
  atomicAdd(&gout[o + 1], v.y);
  atomicAdd(&gout[o + 2], v.z);
  atomicAdd(&gout[o + 3], v.w);
  if (c == 0) atomicAdd(&gcnt[g], 1.0f);
}

// out[g] = fc_b + sum_c (gout[g][c]/max(cnt,1)) * fc_w[c]
__global__ void k_fc(const float* __restrict__ gout, const float* __restrict__ gcnt,
                     const float* __restrict__ w, const float* __restrict__ b,
                     float* __restrict__ out)
{
  __shared__ float red[HH];
  int g = blockIdx.x, c = threadIdx.x;
  float cnt = fmaxf(gcnt[g], 1.0f);
  red[c] = (gout[(size_t)g * HH + c] / cnt) * w[c];
  __syncthreads();
  for (int s = HH >> 1; s > 0; s >>= 1) {
    if (c < s) red[c] += red[c + s];
    __syncthreads();
  }
  if (c == 0) out[g] = red[0] + b[0];
}

// ---------------------------------------------------------------------------
// Host side
// ---------------------------------------------------------------------------
struct LinP { const float* w; const float* b; };
struct BnP  { const float* g; const float* b; };
struct MlpP { LinP lin; BnP bn; };
struct EggcP {
  LinP src_gate, dst_gate, edge_gate, src_update, dst_update;
  BnP bn_nodes, bn_edges;
};

static EggcP eggc_at(void* const* din, int b0) {
  auto F = [&](int i) { return (const float*)din[b0 + i]; };
  EggcP p;
  p.bn_edges   = { F(1),  F(0)  };
  p.bn_nodes   = { F(3),  F(2)  };
  p.dst_gate   = { F(5),  F(4)  };
  p.dst_update = { F(7),  F(6)  };
  p.edge_gate  = { F(9),  F(8)  };
  p.src_gate   = { F(11), F(10) };
  p.src_update = { F(13), F(12) };
  return p;
}
static MlpP mlp_at(void* const* din, int b0) {
  auto F = [&](int i) { return (const float*)din[b0 + i]; };
  MlpP m;
  m.bn  = { F(1), F(0) };
  m.lin = { F(3), F(2) };
  return m;
}

static inline unsigned gb(long n) { return (unsigned)((n + 255) / 256); }

static void gemm(hipStream_t s, const float* A, const float* W, const float* b,
                 float* C, long rows, int K, int Ncols) {
  dim3 grid((unsigned)(Ncols / TN), (unsigned)((rows + TM - 1) / TM));
  k_gemm_bf16<<<grid, dim3(256), 0, s>>>(A, W, b, C, (int)rows, K, Ncols);
}
static void zero(hipStream_t s, float* p, long n) {
  k_zero<<<gb(n), 256, 0, s>>>(p, n);
}
static void bn_silu(hipStream_t s, const float* t, const float* res, float* dst,
                    const BnP& bn, float* stats, long rows, int C) {
  zero(s, stats, 2L * C);
  k_bn_stats<<<dim3(512), dim3(C), 0, s>>>(t, rows, C, stats);
  k_bn_act<<<gb(rows * C), 256, 0, s>>>(t, res, dst, bn.g, bn.b, stats, rows, C - 1);
}

// One EGGC layer. x updated in place (nx rows); y_out = y_in + silu(bn(m))
// (ne rows; y_out may alias y_in). Scratch: TA/TB/TD >= nx*H, TC >= ne*H.
static void run_eggc(hipStream_t s, const EggcP& p, const int* src, const int* dst,
                     float* x, long nx, const float* y_in, float* y_out, long ne,
                     float* TA, float* TB, float* TC, float* TD, float* STATS)
{
  gemm(s, x, p.src_gate.w, p.src_gate.b, TA, nx, HH, HH);
  gemm(s, x, p.dst_gate.w, p.dst_gate.b, TB, nx, HH, HH);
  gemm(s, y_in, p.edge_gate.w, p.edge_gate.b, TC, ne, HH, HH);
  k_gather_add<<<gb(ne * C4), 256, 0, s>>>((float4*)TC, (const float4*)TA,
                                           (const float4*)TB, src, dst, ne);
  gemm(s, x, p.dst_update.w, p.dst_update.b, TD, nx, HH, HH);  // Bh
  zero(s, TA, nx * HH);
  zero(s, TB, nx * HH);
  k_scatter_gate<<<gb(ne * C4), 256, 0, s>>>((const float4*)TC, (const float4*)TD,
                                             src, dst, TA, TB, ne);
  gemm(s, x, p.src_update.w, p.src_update.b, TD, nx, HH, HH);  // Au (overwrite Bh)
  k_combine<<<gb(nx * HH), 256, 0, s>>>(TD, TA, TB, nx * HH);
  bn_silu(s, TD, x, x, p.bn_nodes, STATS, nx, HH);        // x += silu(bn(Au+h))
  bn_silu(s, TC, y_in, y_out, p.bn_edges, STATS, ne, HH); // y' = y + silu(bn(m))
}

extern "C" void kernel_launch(void* const* d_in, const int* in_sizes, int n_in,
                              void* d_out, int out_size, void* d_ws, size_t ws_size,
                              hipStream_t stream)
{
  (void)in_sizes; (void)n_in; (void)out_size; (void)ws_size;

  const float* atom  = (const float*)d_in[0];
  const float* r     = (const float*)d_in[1];
  const float* angle = (const float*)d_in[2];
  const int* g_src  = (const int*)d_in[193];
  const int* g_dst  = (const int*)d_in[194];
  const int* lg_src = (const int*)d_in[195];
  const int* lg_dst = (const int*)d_in[196];
  const int* ngraph = (const int*)d_in[197];

  // params (see index map at top of file)
  EggcP alignn_edge[4], alignn_node[4], gcn[4];
  for (int i = 0; i < 4; ++i) {
    alignn_edge[i] = eggc_at(d_in, 3 + 28 * i);
    alignn_node[i] = eggc_at(d_in, 3 + 28 * i + 14);
    gcn[i]         = eggc_at(d_in, 137 + 14 * i);
  }
  MlpP angle_emb1 = mlp_at(d_in, 115);
  MlpP angle_emb2 = mlp_at(d_in, 119);
  MlpP atom_emb   = mlp_at(d_in, 123);
  MlpP edge_emb1  = mlp_at(d_in, 127);
  MlpP edge_emb2  = mlp_at(d_in, 131);
  const float* fc_b = (const float*)d_in[135];
  const float* fc_w = (const float*)d_in[136];

  // Workspace layout (floats). Total ~340.6M floats ~= 1.37 GB.
  const size_t NH = (size_t)NN * HH;
  const size_t EH = (size_t)NE * HH;
  const size_t TH = (size_t)NT * HH;
  float* ws = (float*)d_ws;
  float* X     = ws;             // N x H
  float* Y     = X + NH;         // E x H
  float* Z     = Y + EH;         // T x H
  float* M     = Z + TH;         // E x H
  float* TA    = M + EH;         // E x H scratch / sum_sh
  float* TB    = TA + EH;        // E x H scratch / sum_s
  float* TC    = TB + EH;        // T x H scratch (m / angle path)
  float* TD    = TC + TH;        // E x H scratch (Bh / Au / angle hidden)
  float* STATS = TD + EH;        // 2*H
  float* GOUT  = STATS + 2 * HH; // G x H
  float* GCNT  = GOUT + (size_t)NG * HH; // G

  // ---- embeddings ----
  // atom: X = silu(bn(atom @ W + b))
  gemm(stream, atom, atom_emb.lin.w, atom_emb.lin.b, X, NN, 92, HH);
  bn_silu(stream, X, nullptr, X, atom_emb.bn, STATS, NN, HH);

  // edge: rbf(|r|) -> 80 -> 64 -> 256
  k_edge_rbf<<<gb(NE), 256, 0, stream>>>(r, TA, NE);
  gemm(stream, TA, edge_emb1.lin.w, edge_emb1.lin.b, TB, NE, 80, 64);
  bn_silu(stream, TB, nullptr, TB, edge_emb1.bn, STATS, NE, 64);
  gemm(stream, TB, edge_emb2.lin.w, edge_emb2.lin.b, Y, NE, 64, HH);
  bn_silu(stream, Y, nullptr, Y, edge_emb2.bn, STATS, NE, HH);

  // angle: rbf -> 40 -> 64 -> 256
  k_angle_rbf<<<gb(NT), 256, 0, stream>>>(angle, TC, NT);
  gemm(stream, TC, angle_emb1.lin.w, angle_emb1.lin.b, TD, NT, 40, 64);
  bn_silu(stream, TD, nullptr, TD, angle_emb1.bn, STATS, NT, 64);
  gemm(stream, TD, angle_emb2.lin.w, angle_emb2.lin.b, Z, NT, 64, HH);
  bn_silu(stream, Z, nullptr, Z, angle_emb2.bn, STATS, NT, HH);

  // ---- ALIGNN layers ----
  float* Ybuf = Y;
  float* Mbuf = M;
  for (int i = 0; i < 4; ++i) {
    // node eggc over atom graph: updates X, writes y' into Mbuf
    run_eggc(stream, alignn_node[i], g_src, g_dst, X, NN, Ybuf, Mbuf, NE,
             TA, TB, TC, TD, STATS);
    // edge eggc over line graph: x-role = Mbuf (E rows), y = Z (T rows)
    run_eggc(stream, alignn_edge[i], lg_src, lg_dst, Mbuf, NE, Z, Z, NT,
             TA, TB, TC, TD, STATS);
    float* t = Ybuf; Ybuf = Mbuf; Mbuf = t;  // Ybuf now holds updated edge feats
  }
  // ---- GCN layers ----
  for (int i = 0; i < 4; ++i) {
    run_eggc(stream, gcn[i], g_src, g_dst, X, NN, Ybuf, Ybuf, NE,
             TA, TB, TC, TD, STATS);
  }

  // ---- readout ----
  zero(stream, GOUT, (long)NG * HH);
  zero(stream, GCNT, NG);
  k_pool<<<gb((long)NN * C4), 256, 0, stream>>>(
      (const float4*)X, ngraph, GOUT, GCNT, NN);
  k_fc<<<dim3(NG), dim3(HH), 0, stream>>>(GOUT, GCNT, fc_w, fc_b,
                                          (float*)d_out);
}
